// LSTMModel_85899345986
// MI455X (gfx1250) — matrix operand order repaired and checked by reading, not compiled, not run on hardware
//
#include <hip/hip_runtime.h>
#include <cstdint>
#include <cstddef>

typedef __attribute__((ext_vector_type(16))) __bf16 v16bf;
typedef __attribute__((ext_vector_type(8)))  __bf16 v8bf;
typedef __attribute__((ext_vector_type(8)))  float  v8f;

namespace {

constexpr int kBatch        = 65536;
constexpr int kT            = 30;
constexpr int kI0           = 17;
constexpr int kH            = 50;
constexpr int kNG           = 200;   // 4*H gate columns
constexpr int kNPad         = 208;   // 13 * 16
constexpr int kKPad         = 128;   // max K (100) padded to 4 WMMA k-steps
constexpr int kNT           = 13;
constexpr int kWaves        = 4;
constexpr int kRowsPerWave  = 16;
constexpr int kRowsPerBlock = kWaves * kRowsPerWave;  // 64
constexpr int kThreads      = kWaves * 32;            // 128

// LDS layout (bytes)
constexpr int    kSWElems = 3 * kNPad * kKPad;                   // bf16 weights, 3 layers
constexpr int    kSBElems = 3 * kNPad;                           // fused biases (f32)
constexpr int    kSAElems = kWaves * 3 * kRowsPerWave * kKPad;   // A tiles (bf16)
constexpr int    kSGElems = kWaves * kRowsPerWave * kNPad;       // gate buffer (f32)
constexpr size_t kOffW = 0;
constexpr size_t kOffB = kOffW + sizeof(unsigned short) * (size_t)kSWElems;  // 159744
constexpr size_t kOffA = kOffB + sizeof(float) * (size_t)kSBElems;           // 162240 (32B-aligned)
constexpr size_t kOffG = kOffA + sizeof(unsigned short) * (size_t)kSAElems;  // 211392
constexpr size_t kSmemBytes = kOffG + sizeof(float) * (size_t)kSGElems;      // 264640 < 320KB

__device__ __forceinline__ unsigned short f2bf(float f) {  // round-to-nearest-even
  unsigned u = __float_as_uint(f);
  u += 0x7fffu + ((u >> 16) & 1u);
  return (unsigned short)(u >> 16);
}
__device__ __forceinline__ float bf2f(unsigned short h) {
  return __uint_as_float(((unsigned)h) << 16);
}
__device__ __forceinline__ float sigmoidf_fast(float x) {
  return 1.0f / (1.0f + __expf(-x));
}
__device__ __forceinline__ float tanhf_fast(float x) {
  return 2.0f / (1.0f + __expf(-2.0f * x)) - 1.0f;
}

// One LSTM layer, one timestep, for one wave's 16-row batch tile.
// A tile layout per layer: [16 rows][kKPad] bf16 = [input (idim) | h_own (kH) | zero pad]
template<int KSTEPS, int KOFF, bool LAST>
__device__ __forceinline__ void layer_step(
    const unsigned short* __restrict__ Wl,  // [kNPad][kKPad] bf16: W[n][k] = [W_ih | W_hh]
    const float*          __restrict__ Bl,  // [kNPad] fused bias b_ih + b_hh
    unsigned short*       __restrict__ Al,  // this layer's A tile
    unsigned short*       __restrict__ An,  // next layer's A tile (ignored if LAST)
    float*                __restrict__ Gw,  // [16][kNPad] gate scratch (f32)
    float (&cst)[25], int lane)
{
  const int aRow = lane & 15;
  const int kh   = lane >> 4;       // 0 or 1
  // --- A fragments (shared across all 13 N tiles) ---
  v16bf af[KSTEPS];
#pragma unroll
  for (int ks = 0; ks < KSTEPS; ++ks) {
    // 16-bit A 16x32 layout: lane holds K = base..base+7 and base+16..base+23, base = ks*32 + kh*8
    const unsigned short* p = Al + aRow * kKPad + ks * 32 + kh * 8;
    v8bf lo = *(const v8bf*)p;
    v8bf hi = *(const v8bf*)(p + 16);
    af[ks] = __builtin_shufflevector(lo, hi, 0,1,2,3,4,5,6,7,8,9,10,11,12,13,14,15);
  }
  const int ncLane = lane & 15;
  const int rb     = kh * 8;
  // --- 13 N-tiles of the gate GEMM ---
  for (int nt = 0; nt < kNT; ++nt) {
    const int ncol = nt * 16 + ncLane;
    const float bv = Bl[ncol];                       // bias folded into accumulator
    v8f acc = {bv, bv, bv, bv, bv, bv, bv, bv};
#pragma unroll
    for (int ks = 0; ks < KSTEPS; ++ks) {
      // 16-bit B 32x16 layout: lane holds col ncLane, K = base..base+7 and base+16..base+23, base = ks*32 + kh*8
      const unsigned short* q = Wl + ncol * kKPad + ks * 32 + kh * 8;
      v8bf blo = *(const v8bf*)q;
      v8bf bhi = *(const v8bf*)(q + 16);
      v16bf bfrag = __builtin_shufflevector(blo, bhi, 0,1,2,3,4,5,6,7,8,9,10,11,12,13,14,15);
      acc = __builtin_amdgcn_wmma_f32_16x16x32_bf16(
          /*neg_a=*/false, af[ks], /*neg_b=*/false, bfrag,
          /*c_mod=*/(short)0, acc, /*reuse_a=*/false, /*reuse_b=*/false);
    }
    // C/D layout: VGPR v -> row v + 8*(lane>>4), col = ncol
#pragma unroll
    for (int v = 0; v < 8; ++v)
      Gw[(rb + v) * kNPad + ncol] = acc[v];
  }
  // --- LSTM cell: 16 rows x 50 units = 800 values, 25 per lane ---
#pragma unroll
  for (int j = 0; j < 25; ++j) {
    const int flat = j * 32 + lane;
    const int r = flat / kH;
    const int u = flat - r * kH;
    const float* g = Gw + r * kNPad;
    const float ig = sigmoidf_fast(g[u]);
    const float fg = sigmoidf_fast(g[kH + u]);
    const float gg = tanhf_fast(g[2 * kH + u]);
    const float og = sigmoidf_fast(g[3 * kH + u]);
    const float c  = fg * cst[j] + ig * gg;
    cst[j] = c;
    const float h = og * tanhf_fast(c);
    const unsigned short hb = f2bf(h);
    Al[r * kKPad + KOFF + u] = hb;        // own h_{t-1} slot for next timestep
    if (!LAST) An[r * kKPad + u] = hb;    // next layer's input at this timestep
  }
}

__global__ __launch_bounds__(kThreads)
void lstm3_fused_kernel(
    const float* __restrict__ x,
    const float* __restrict__ wih0, const float* __restrict__ whh0,
    const float* __restrict__ bih0, const float* __restrict__ bhh0,
    const float* __restrict__ wih1, const float* __restrict__ whh1,
    const float* __restrict__ bih1, const float* __restrict__ bhh1,
    const float* __restrict__ wih2, const float* __restrict__ whh2,
    const float* __restrict__ bih2, const float* __restrict__ bhh2,
    const float* __restrict__ fcw,  const float* __restrict__ fcb,
    float* __restrict__ out)
{
  extern __shared__ __attribute__((aligned(32))) char smem[];
  unsigned short* sW = (unsigned short*)(smem + kOffW);
  float*          sB = (float*)(smem + kOffB);
  unsigned short* sA = (unsigned short*)(smem + kOffA);
  float*          sG = (float*)(smem + kOffG);

  const int tid  = threadIdx.x;
  const int lane = tid & 31;
  const int wave = tid >> 5;

  // ---- cooperative fill: concatenated weights [W_ih | W_hh] as bf16, fused biases ----
  {
    const float* wihs[3] = {wih0, wih1, wih2};
    const float* whhs[3] = {whh0, whh1, whh2};
    const float* bihs[3] = {bih0, bih1, bih2};
    const float* bhhs[3] = {bhh0, bhh1, bhh2};
#pragma unroll
    for (int l = 0; l < 3; ++l) {
      const int idim = (l == 0) ? kI0 : kH;
      const float* wih = wihs[l];
      const float* whh = whhs[l];
      unsigned short* W = sW + l * kNPad * kKPad;
      for (int idx = tid; idx < kNPad * kKPad; idx += kThreads) {
        const int n = idx / kKPad;
        const int k = idx - n * kKPad;
        float v = 0.f;
        if (n < kNG) {
          if (k < idim)           v = wih[n * idim + k];
          else if (k < idim + kH) v = whh[n * kH + (k - idim)];
        }
        W[idx] = f2bf(v);
      }
      for (int n = tid; n < kNPad; n += kThreads)
        sB[l * kNPad + n] = (n < kNG) ? (bihs[l][n] + bhhs[l][n]) : 0.f;
    }
    for (int idx = tid; idx < kSAElems; idx += kThreads) sA[idx] = 0;  // h0 = 0, K pad = 0
  }
  __syncthreads();

  unsigned short* A0 = sA + (wave * 3 + 0) * kRowsPerWave * kKPad;
  unsigned short* A1 = sA + (wave * 3 + 1) * kRowsPerWave * kKPad;
  unsigned short* A2 = sA + (wave * 3 + 2) * kRowsPerWave * kKPad;
  float* Gw = sG + wave * kRowsPerWave * kNPad;
  const int rowBase = blockIdx.x * kRowsPerBlock + wave * kRowsPerWave;

  float cst0[25], cst1[25], cst2[25];
#pragma unroll
  for (int j = 0; j < 25; ++j) { cst0[j] = 0.f; cst1[j] = 0.f; cst2[j] = 0.f; }

  for (int t = 0; t < kT; ++t) {
    // stage x[:, t, :] (fp32 -> bf16) into layer-0 A tile columns [0, 17)
    for (int j = lane; j < kRowsPerWave * kI0; j += 32) {
      const int r = j / kI0;
      const int i = j - r * kI0;
      const size_t gx = ((size_t)(rowBase + r) * kT + (size_t)t) * kI0 + i;
      A0[r * kKPad + i] = f2bf(x[gx]);
      if (t + 1 < kT) __builtin_prefetch(&x[gx + kI0], 0, 1);  // next timestep
    }
    // layer pipeline inside the timestep: no hidden-state sequences ever staged
    layer_step<3, kI0, false>(sW + 0 * kNPad * kKPad, sB + 0 * kNPad, A0, A1, Gw, cst0, lane);
    layer_step<4, kH,  false>(sW + 1 * kNPad * kKPad, sB + 1 * kNPad, A1, A2, Gw, cst1, lane);
    layer_step<4, kH,  true >(sW + 2 * kNPad * kKPad, sB + 2 * kNPad, A2, A2, Gw, cst2, lane);
  }

  // final FC(50->1) + ReLU on layer-2 h at t = T-1 (lives at A2[:, kH..kH+50))
  if (lane < kRowsPerWave) {
    const unsigned short* h2 = A2 + lane * kKPad + kH;
    float s = fcb[0];
#pragma unroll
    for (int u = 0; u < kH; ++u) s += bf2f(h2[u]) * fcw[u];
    out[rowBase + lane] = fmaxf(s, 0.f);
  }
}

}  // namespace

extern "C" void kernel_launch(void* const* d_in, const int* in_sizes, int n_in,
                              void* d_out, int out_size, void* d_ws, size_t ws_size,
                              hipStream_t stream) {
  (void)in_sizes; (void)n_in; (void)out_size; (void)d_ws; (void)ws_size;
  const float* x    = (const float*)d_in[0];
  const float* wih0 = (const float*)d_in[1];
  const float* whh0 = (const float*)d_in[2];
  const float* bih0 = (const float*)d_in[3];
  const float* bhh0 = (const float*)d_in[4];
  const float* wih1 = (const float*)d_in[5];
  const float* whh1 = (const float*)d_in[6];
  const float* bih1 = (const float*)d_in[7];
  const float* bhh1 = (const float*)d_in[8];
  const float* wih2 = (const float*)d_in[9];
  const float* whh2 = (const float*)d_in[10];
  const float* bih2 = (const float*)d_in[11];
  const float* bhh2 = (const float*)d_in[12];
  const float* fcw  = (const float*)d_in[13];
  const float* fcb  = (const float*)d_in[14];
  float* out = (float*)d_out;

  dim3 grid(kBatch / kRowsPerBlock);   // 1024 blocks of 64 batch rows
  dim3 block(kThreads);                // 4 waves, one 16-row WMMA tile each
  lstm3_fused_kernel<<<grid, block, kSmemBytes, stream>>>(
      x, wih0, whh0, bih0, bhh0, wih1, whh1, bih1, bhh1,
      wih2, whh2, bih2, bhh2, fcw, fcb, out);
}
